// QuantHGTCavAttention_15899968930299
// MI455X (gfx1250) — matrix operand order repaired and checked by reading, not compiled, not run on hardware
//
#include <hip/hip_runtime.h>
#include <hip/hip_bf16.h>

// ---------------------------------------------------------------------------
// QuantHGTCavAttention for MI455X (gfx1250, wave32, WMMA + async-to-LDS)
// B=2, L=5, H=32, W=100 (HW=3200), C=D=256, HEADS=8, DH=32, T=2
// ---------------------------------------------------------------------------

typedef _Float16 half8  __attribute__((ext_vector_type(8)));
typedef _Float16 half16 __attribute__((ext_vector_type(16)));
typedef float    float8 __attribute__((ext_vector_type(8)));

#define HGT_B      2
#define HGT_L      5
#define HGT_HW     3200
#define HGT_C      256
#define HGT_HEADS  8
#define HGT_DH     32
#define HGT_SCALE  0.17677669529663687f   // 32^-0.5

#define LDA_F32 36   // A tile LDS row stride in floats (32 data + 4 pad, 16B aligned)
#define LDB_F16 40   // B tile LDS row stride in halfs  (32 data + 8 pad, 16B aligned)

// CDNA5 async copy: global -> LDS, 16B per lane, tracked by ASYNCcnt.
// IOFFSET is added to BOTH the LDS and the global address (ISA 15.18.3), so the
// four 16B-apart copies share one LDS-address VGPR and one global addr pair.
__device__ __forceinline__ void async_load_4x_b128(unsigned lds_off, const float* gaddr)
{
    asm volatile("global_load_async_to_lds_b128 %0, %1, off\n\t"
                 "global_load_async_to_lds_b128 %0, %1, off offset:16\n\t"
                 "global_load_async_to_lds_b128 %0, %1, off offset:32\n\t"
                 "global_load_async_to_lds_b128 %0, %1, off offset:48"
                 :: "v"(lds_off), "v"(gaddr) : "memory");
}
__device__ __forceinline__ void wait_asynccnt0()
{
    asm volatile("s_wait_asynccnt 0" ::: "memory");
}

// ---------------------------------------------------------------------------
// Typed GEMM:  Out[bl*3200+row][n] = sum_k A[bl*3200+row][k] * W[type(bl)][k][n]
//                                    + bias[type(bl)][n]
// A: [10*3200 x 256] f32 row-major; W: [T,256,256]; bias: [T,256]
// blockIdx.z selects (which output) * 4 + (ntile): q/k/v fused in one launch.
// grid = (10, 50, 12|4); block = 128 (4 waves, 2x2 wave grid, 32x32 per wave)
// ---------------------------------------------------------------------------
__global__ __launch_bounds__(128)
void hgt_typed_gemm(const float* __restrict__ A,
                    const float* __restrict__ W0, const float* __restrict__ W1,
                    const float* __restrict__ W2,
                    const float* __restrict__ b0, const float* __restrict__ b1,
                    const float* __restrict__ b2,
                    const float* __restrict__ prior,
                    float* __restrict__ O0, float* __restrict__ O1,
                    float* __restrict__ O2)
{
    __shared__ float    lA[64 * LDA_F32];   // A tile: 64 rows (M) x 32 (K), f32 (async fill)
    __shared__ _Float16 lB[64 * LDB_F16];   // B tile transposed: 64 rows (N) x 32 (K), f16

    const int bl    = blockIdx.x;            // b*L + l
    const int mtile = blockIdx.y;            // 0..49
    const int which = blockIdx.z >> 2;       // 0:q 1:k 2:v   (stage 3: always 0)
    const int ntile = blockIdx.z & 3;        // 0..3
    const int typ   = (int)prior[bl * HGT_HW * 3 + 2];   // prior_encoding[b,l,0,0,2]

    const float* Wt   = (which == 0) ? W0 : (which == 1) ? W1 : W2;
    const float* bias = (which == 0) ? b0 : (which == 1) ? b1 : b2;
    float*       Out  = (which == 0) ? O0 : (which == 1) ? O1 : O2;

    const int t    = threadIdx.x;
    const int lane = t & 31;
    const int wave = t >> 5;
    const int wr   = wave >> 1;              // wave row (0..1)
    const int wc   = wave & 1;               // wave col (0..1)

    // cooperative A loader: thread -> (row, 16-float half of the 32-wide K slab)
    const int rowA  = t >> 1;                // 0..63
    const int halfA = (t & 1) * 16;          // 0 or 16
    const float* gA = A + (size_t)(bl * HGT_HW + mtile * 64 + rowA) * HGT_C + halfA;
    const unsigned ldsA =
        (unsigned)(uintptr_t)(lA + rowA * LDA_F32 + halfA);  // LDS byte offset

    // cooperative B loader: thread -> (k row, 16 consecutive n)
    const int kB = t >> 2;                   // 0..31
    const int nB = (t & 3) * 16;             // 0,16,32,48
    const float* gB = Wt + (size_t)(typ * HGT_C + kB) * HGT_C + ntile * 64 + nB;

    float8 acc[2][2] = {};

    for (int k0 = 0; k0 < HGT_C; k0 += 32) {
        // ---- A tile: raw f32 via CDNA5 async copies (ASYNCcnt path) -------
        async_load_4x_b128(ldsA, gA + k0);

        // ---- B tile transposed (f32 -> f16, manual: async can't transpose)
        const float4* pb = (const float4*)(gB + (size_t)k0 * HGT_C);
        #pragma unroll
        for (int q4 = 0; q4 < 4; ++q4) {
            float4 f = pb[q4];
            lB[(nB + q4 * 4 + 0) * LDB_F16 + kB] = (_Float16)f.x;
            lB[(nB + q4 * 4 + 1) * LDB_F16 + kB] = (_Float16)f.y;
            lB[(nB + q4 * 4 + 2) * LDB_F16 + kB] = (_Float16)f.z;
            lB[(nB + q4 * 4 + 3) * LDB_F16 + kB] = (_Float16)f.w;
        }
        if (k0 + 32 < HGT_C)   // prefetch next B slab (global_prefetch_b8)
            __builtin_prefetch(gB + (size_t)(k0 + 32) * HGT_C, 0, 0);

        wait_asynccnt0();
        __syncthreads();

        // ---- assemble fragments per documented gfx1250 layouts ------------
        // A (16-bit 16x32): lane<16 -> K {0..7,16..23}; lane>=16 -> K {8..15,24..31}
        half16 af[2];
        #pragma unroll
        for (int rt = 0; rt < 2; ++rt) {
            int row = wr * 32 + rt * 16 + (lane & 15);
            const float* base = lA + row * LDA_F32 + ((lane & 16) ? 8 : 0);
            float4 f0 = ((const float4*)base)[0];
            float4 f1 = ((const float4*)base)[1];
            float4 f2 = ((const float4*)(base + 16))[0];
            float4 f3 = ((const float4*)(base + 16))[1];
            af[rt][0]  = (_Float16)f0.x;  af[rt][1]  = (_Float16)f0.y;
            af[rt][2]  = (_Float16)f0.z;  af[rt][3]  = (_Float16)f0.w;
            af[rt][4]  = (_Float16)f1.x;  af[rt][5]  = (_Float16)f1.y;
            af[rt][6]  = (_Float16)f1.z;  af[rt][7]  = (_Float16)f1.w;
            af[rt][8]  = (_Float16)f2.x;  af[rt][9]  = (_Float16)f2.y;
            af[rt][10] = (_Float16)f2.z;  af[rt][11] = (_Float16)f2.w;
            af[rt][12] = (_Float16)f3.x;  af[rt][13] = (_Float16)f3.y;
            af[rt][14] = (_Float16)f3.z;  af[rt][15] = (_Float16)f3.w;
        }
        // B (16-bit 32x16): lane<16 -> col n=lane, K {0..7,16..23}; lane>=16 -> K {8..15,24..31}
        half16 bf[2];
        #pragma unroll
        for (int ct = 0; ct < 2; ++ct) {
            int col = wc * 32 + ct * 16 + (lane & 15);
            const _Float16* base = lB + col * LDB_F16 + ((lane & 16) ? 8 : 0);
            half8 lo = *(const half8*)(base);
            half8 hi = *(const half8*)(base + 16);
            #pragma unroll
            for (int e = 0; e < 8; ++e) { bf[ct][e] = lo[e]; bf[ct][e + 8] = hi[e]; }
        }

        #pragma unroll
        for (int rt = 0; rt < 2; ++rt)
            #pragma unroll
            for (int ct = 0; ct < 2; ++ct)
                acc[rt][ct] = __builtin_amdgcn_wmma_f32_16x16x32_f16(
                    false, af[rt], false, bf[ct], (short)0, acc[rt][ct], false, false);

        __syncthreads();
    }

    // ---- epilogue: bias + store (C/D layout: vgpr i -> M=i / i+8) ---------
    #pragma unroll
    for (int rt = 0; rt < 2; ++rt) {
        #pragma unroll
        for (int ct = 0; ct < 2; ++ct) {
            int col  = ntile * 64 + wc * 32 + ct * 16 + (lane & 15);
            float bv = bias[typ * HGT_C + col];
            int row0 = mtile * 64 + wr * 32 + rt * 16 + ((lane & 16) ? 8 : 0);
            #pragma unroll
            for (int r = 0; r < 8; ++r)
                Out[(size_t)(bl * HGT_HW + row0 + r) * HGT_C + col] = acc[rt][ct][r] + bv;
        }
    }
}

// ---------------------------------------------------------------------------
// Attention core: one wave per (b, head m, pixel hw).
//   att[i,j] = softmax_j( SCALE * q_i^T * W_att[rel(i,j),m] * k_j , mask )
//   out[i]   = sum_j att[i,j] * (v_j^T W_msg[rel(i,j),m])
// Q/K/V/O layout: [B, L, HW, 256] with column = m*32 + dh
// grid = (B*HEADS, HW/8); block = 256 (8 waves)
// ---------------------------------------------------------------------------
__global__ __launch_bounds__(256)
void hgt_attention(const float* __restrict__ Q, const float* __restrict__ K,
                   const float* __restrict__ V, const float* __restrict__ mask,
                   const float* __restrict__ prior,
                   const float* __restrict__ rel_att, const float* __restrict__ rel_msg,
                   float* __restrict__ O)
{
    __shared__ float latt[4 * 32 * 33];   // [rel][p][q], padded
    __shared__ float lmsg[4 * 32 * 33];   // [rel][p][c], padded

    const int b = blockIdx.x >> 3;
    const int m = blockIdx.x & 7;

    // all 4 relation matrices for this head into LDS (pad 33 -> conflict-free)
    for (int idx = threadIdx.x; idx < 4 * 32 * 32; idx += 256) {
        int r = idx >> 10;
        int a = (idx >> 5) & 31;
        int c = idx & 31;
        int g = ((r * HGT_HEADS + m) * 32 + a) * 32 + c;
        latt[r * (32 * 33) + a * 33 + c] = rel_att[g];
        lmsg[r * (32 * 33) + a * 33 + c] = rel_msg[g];
    }

    int tb[HGT_L];
    #pragma unroll
    for (int l = 0; l < HGT_L; ++l)
        tb[l] = (int)prior[(size_t)(b * HGT_L + l) * HGT_HW * 3 + 2];

    const int wave = threadIdx.x >> 5;
    const int lane = threadIdx.x & 31;
    const int hw   = blockIdx.y * 8 + wave;

    float qr[HGT_L], kr[HGT_L], vr[HGT_L];
    #pragma unroll
    for (int i = 0; i < HGT_L; ++i) {
        size_t base = (size_t)((b * HGT_L + i) * HGT_HW + hw) * HGT_C + m * HGT_DH + lane;
        qr[i] = Q[base]; kr[i] = K[base]; vr[i] = V[base];
    }
    float mk = (lane < 25) ? mask[(size_t)(b * HGT_HW + hw) * 25 + lane] : 1.0f;

    __syncthreads();

    // t1[j][ti][lane=p] = sum_q W_att[ti*2+tj(j)][p][q] * k_j[q]
    float t1[HGT_L][2];
    #pragma unroll
    for (int j = 0; j < HGT_L; ++j)
        #pragma unroll
        for (int ti = 0; ti < 2; ++ti) {
            const float* wrow = latt + (ti * 2 + tb[j]) * (32 * 33) + lane * 33;
            float s = 0.0f;
            #pragma unroll
            for (int p = 0; p < 32; ++p) s += wrow[p] * __shfl(kr[j], p, 32);
            t1[j][ti] = s;
        }

    // att[i][j] (uniform across wave after xor-reduction)
    float att[HGT_L][HGT_L];
    #pragma unroll
    for (int i = 0; i < HGT_L; ++i)
        #pragma unroll
        for (int j = 0; j < HGT_L; ++j) {
            float s = qr[i] * t1[j][tb[i]];
            #pragma unroll
            for (int off = 16; off > 0; off >>= 1) s += __shfl_xor(s, off, 32);
            s *= HGT_SCALE;
            float mv = __shfl(mk, i * 5 + j, 32);
            att[i][j] = (mv == 0.0f) ? -__builtin_inff() : s;
        }

    // softmax over j
    #pragma unroll
    for (int i = 0; i < HGT_L; ++i) {
        float mx = att[i][0];
        #pragma unroll
        for (int j = 1; j < HGT_L; ++j) mx = fmaxf(mx, att[i][j]);
        float sum = 0.0f;
        #pragma unroll
        for (int j = 0; j < HGT_L; ++j) { float e = __expf(att[i][j] - mx); att[i][j] = e; sum += e; }
        float inv = 1.0f / sum;
        #pragma unroll
        for (int j = 0; j < HGT_L; ++j) att[i][j] *= inv;
    }

    // t2[j][ti][lane=c] = sum_p v_j[p] * W_msg[ti*2+tj(j)][p][c]
    float t2[HGT_L][2];
    #pragma unroll
    for (int j = 0; j < HGT_L; ++j)
        #pragma unroll
        for (int ti = 0; ti < 2; ++ti) {
            const float* wmat = lmsg + (ti * 2 + tb[j]) * (32 * 33);
            float s = 0.0f;
            #pragma unroll
            for (int p = 0; p < 32; ++p) s += __shfl(vr[j], p, 32) * wmat[p * 33 + lane];
            t2[j][ti] = s;
        }

    #pragma unroll
    for (int i = 0; i < HGT_L; ++i) {
        float o = 0.0f;
        #pragma unroll
        for (int j = 0; j < HGT_L; ++j) o += att[i][j] * t2[j][tb[i]];
        O[(size_t)((b * HGT_L + i) * HGT_HW + hw) * HGT_C + m * HGT_DH + lane] = o;
    }
}

// ---------------------------------------------------------------------------
// Launch: fused q/k/v typed GEMM -> attention -> output typed GEMM
// Workspace: Q,K,V,O each B*L*HW*256 f32 = 32 MB (128 MB total), fully
// overwritten each call (deterministic, poison-safe).
// ---------------------------------------------------------------------------
extern "C" void kernel_launch(void* const* d_in, const int* in_sizes, int n_in,
                              void* d_out, int out_size, void* d_ws, size_t ws_size,
                              hipStream_t stream)
{
    const float* x       = (const float*)d_in[0];
    const float* mask    = (const float*)d_in[1];
    const float* prior   = (const float*)d_in[2];
    const float* Wq      = (const float*)d_in[3];
    const float* bq      = (const float*)d_in[4];
    const float* Wk      = (const float*)d_in[5];
    const float* bk      = (const float*)d_in[6];
    const float* Wv      = (const float*)d_in[7];
    const float* bv      = (const float*)d_in[8];
    const float* Wa      = (const float*)d_in[9];
    const float* ba      = (const float*)d_in[10];
    const float* rel_att = (const float*)d_in[11];
    const float* rel_msg = (const float*)d_in[12];
    float* out = (float*)d_out;

    const size_t per = (size_t)HGT_B * HGT_L * HGT_HW * HGT_C;   // 8,192,000 floats
    float* Qb = (float*)d_ws;
    float* Kb = Qb + per;
    float* Vb = Kb + per;
    float* Ob = Vb + per;

    // fused q/k/v projections: x read once, three outputs
    dim3 g1(HGT_B * HGT_L, HGT_HW / 64, 3 * (HGT_C / 64));   // (10, 50, 12)
    hgt_typed_gemm<<<g1, 128, 0, stream>>>(x, Wq, Wk, Wv, bq, bk, bv, prior,
                                           Qb, Kb, Vb);

    hgt_attention<<<dim3(HGT_B * HGT_HEADS, HGT_HW / 8), 256, 0, stream>>>(
        Qb, Kb, Vb, mask, prior, rel_att, rel_msg, Ob);

    // output projection (single weight; selectors all point at Wa/ba/out)
    dim3 g3(HGT_B * HGT_L, HGT_HW / 64, HGT_C / 64);         // (10, 50, 4)
    hgt_typed_gemm<<<g3, 128, 0, stream>>>(Ob, Wa, Wa, Wa, ba, ba, ba, prior,
                                           out, out, out);
}
